// ModulationConvBlock_4561255269281
// MI455X (gfx1250) — matrix operand order repaired and checked by reading, not compiled, not run on hardware
//
#include <hip/hip_runtime.h>

// ---------------------------------------------------------------------------
// Modulated 3x3 conv (StyleGAN2 style) for MI455X / gfx1250, wave32 + WMMA.
// B=8, IN_C=256, OUT_C=256, K=3, H=W=128.
//
//   1) modnorm_kernel : per-(b,oc) demodulation norm  -> invn[2048] (ws)
//   2) modfrag_kernel : demodulated weights in f16, pre-swizzled into the
//                       exact WMMA A-matrix (16x32 f16) lane layout   (ws)
//   3) modconv_kernel : implicit GEMM conv, v_wmma_f32_16x16x32_f16.
//      - wave register-blocks 32oc x 64px (2 A x 4 B -> 8 WMMA per step-4)
//      - one WG covers all 256 oc -> x staged once per strip
//      - double-buffered LDS x tile; next chunk's global loads issued
//        before the 72-WMMA body so they retire under compute
//      - explicit 1-step-ahead B-fragment and 1-tap-ahead A-fragment
//        pipelining with tight live ranges (keeps VGPR count down)
// ---------------------------------------------------------------------------

typedef __attribute__((ext_vector_type(16))) _Float16 v16h;
typedef __attribute__((ext_vector_type(8)))  float    v8f;

#define BB    8
#define HH    128
#define WW    128
#define HW    (HH * WW)
#define KRED  2304              // 9 * 256
#define WSCALE (1.0f / 48.0f)   // 1/sqrt(3*3*256)
#define ACT_SCALE 1.41421356f

#define FRAG_ELEMS (8 * 16 * 8 * 9 * 512)  // b * ocblk * icchunk * tap * 512
#define OB_STRIDE  (8 * 9 * 512)           // halfs between oc-blocks in frag
#define XTILE      (3 * 68 * 40)           // halfs per LDS x tile
#define XLOAD      (3 * 66 * 32)           // elements staged per chunk
#define NLD        25                      // ceil(XLOAD / 256)

// ---------------------------------------------------------------------------
__global__ __launch_bounds__(256)
void modnorm_kernel(const float* __restrict__ weight,
                    const float* __restrict__ code,
                    float* __restrict__ invn) {
  const int bo = blockIdx.x;          // b*256 + oc
  const int b  = bo >> 8;
  const int oc = bo & 255;
  __shared__ float red[256];
  float s = 0.0f;
  for (int j = threadIdx.x; j < KRED; j += 256) {
    const float w  = weight[(size_t)j * 256 + oc] * WSCALE;
    const float cd = code[b * 256 + (j & 255)];
    const float v  = w * cd;
    s += v * v;
  }
  red[threadIdx.x] = s;
  __syncthreads();
  for (int off = 128; off > 0; off >>= 1) {
    if (threadIdx.x < off) red[threadIdx.x] += red[threadIdx.x + off];
    __syncthreads();
  }
  if (threadIdx.x == 0)
    invn[bo] = WSCALE * rsqrtf(red[0] + 1e-8f);
}

// ---------------------------------------------------------------------------
__global__ __launch_bounds__(256)
void modfrag_kernel(const float* __restrict__ weight,
                    const float* __restrict__ code,
                    const float* __restrict__ invn,
                    _Float16* __restrict__ frag) {
  const int idx = blockIdx.x * 256 + threadIdx.x;
  if (idx >= FRAG_ELEMS) return;
  const int f = idx >> 9;
  const int r = idx & 511;
  const int L = r >> 4;
  const int e = r & 15;
  const int tap = f % 9;
  const int g   = f / 9;
  const int c   = g & 7;          // ic chunk
  const int ob  = (g >> 3) & 15;  // oc block of 16
  const int b   = g >> 7;
  const int oc  = ob * 16 + (L & 15);
  const int p   = e >> 1;
  const int tt  = e & 1;
  const int kl  = e + ((L >> 4) << 4);
  const int ic  = c * 32 + kl;
  const float val = weight[(size_t)(tap * 256 + ic) * 256 + oc] *
                    code[b * 256 + ic] * invn[b * 256 + oc];
  frag[idx] = (_Float16)val;
}

// ---------------------------------------------------------------------------
// grid: (256 strips, 8 batches), 256 threads = 8 waves.
// Wave w owns oc blocks {2w, 2w+1} (32 oc) x the full 64-px strip.
// ---------------------------------------------------------------------------
__global__ __launch_bounds__(256)
void modconv_kernel(const float* __restrict__ x,
                    const float* __restrict__ bias,
                    const _Float16* __restrict__ frag,
                    float* __restrict__ out) {
  __shared__ __align__(16) _Float16 ldsx[2 * XTILE];

  const int strip = blockIdx.x;            // 0..255
  const int b     = blockIdx.y;
  const int h     = strip >> 1;
  const int w0    = (strip & 1) << 6;      // 0 or 64
  const int tid   = threadIdx.x;
  const int lane  = tid & 31;
  const int wave  = tid >> 5;              // 0..7
  const int n     = lane & 15;             // N column within 16x16 tile
  const int hi    = lane >> 4;             // 0/1

  v8f acc[2][4] = {};
  float rv[NLD];

  const float* xbat = x + (size_t)b * 256 * HW;
  const _Float16* fbase = frag + (size_t)(b * 16 + wave * 2) * OB_STRIDE;

  // ---- stage chunk 0 into buffer 0
#pragma unroll
  for (int k = 0; k < NLD; ++k) {
    const int i = tid + k * 256;
    float v = 0.0f;
    if (i < XLOAD) {
      const int col = i % 66;
      const int r2  = i / 66;
      const int row = r2 % 3;
      const int ic  = r2 / 3;
      const int gh  = h + row - 1;
      const int gw  = w0 + col - 1;
      if (gh >= 0 && gh < HH && gw >= 0 && gw < WW)
        v = xbat[(size_t)ic * HW + gh * WW + gw];
    }
    rv[k] = v;
  }
#pragma unroll
  for (int k = 0; k < NLD; ++k) {
    const int i = tid + k * 256;
    if (i < XLOAD) {
      const int col = i % 66;
      const int r2  = i / 66;
      const int row = r2 % 3;
      const int ic  = r2 / 3;
      ldsx[(row * 68 + col) * 40 + ic] = (_Float16)rv[k];
    }
  }
  __syncthreads();

  for (int c = 0; c < 8; ++c) {
    // ---- issue next chunk's global loads early; they retire under the WMMAs
    if (c < 7) {
      const float* xb = xbat + (size_t)(c + 1) * 32 * HW;
#pragma unroll
      for (int k = 0; k < NLD; ++k) {
        const int i = tid + k * 256;
        float v = 0.0f;
        if (i < XLOAD) {
          const int col = i % 66;
          const int r2  = i / 66;
          const int row = r2 % 3;
          const int ic  = r2 / 3;
          const int gh  = h + row - 1;
          const int gw  = w0 + col - 1;
          if (gh >= 0 && gh < HH && gw >= 0 && gw < WW)
            v = xb[(size_t)ic * HW + gh * WW + gw];
        }
        rv[k] = v;
      }
    }

    // ---- 72 WMMAs on current buffer, software-pipelined
    const _Float16* cur = ldsx + (c & 1) * XTILE;
    const _Float16* fc  = fbase + (size_t)c * 9 * 512 + lane * 16;
    if (c < 7) __builtin_prefetch((const void*)(fc + 9 * 512), 0, 1);

    // B-fragment address helper (all constants after full unroll):
    //   col = s*16 + n + dw, row = dh, ic half = hi*16
    v16h af0 = *(const v16h*)(fc + 0 * 512);
    v16h af1 = *(const v16h*)(fc + OB_STRIDE + 0 * 512);
    v16h af0n = af0, af1n = af1;
    v16h bf = *(const v16h*)(cur + hi * 16 + (size_t)(n + 0) * 40); // t=0,s=0
    v16h bfn = bf;

#pragma unroll
    for (int step = 0; step < 36; ++step) {
      const int t = step >> 2;
      const int s = step & 3;
      // prefetch next B fragment (one step ahead)
      if (step < 35) {
        const int t2  = (step + 1) >> 2;
        const int s2  = (step + 1) & 3;
        const int dh2 = t2 / 3;
        const int dw2 = t2 % 3;
        bfn = *(const v16h*)(cur + (dh2 * 68) * 40 + hi * 16 +
                             (size_t)(s2 * 16 + n + dw2) * 40);
      }
      // prefetch next tap's A fragments four steps ahead
      if (s == 0 && t < 8) {
        af0n = *(const v16h*)(fc + (t + 1) * 512);
        af1n = *(const v16h*)(fc + OB_STRIDE + (t + 1) * 512);
      }
      acc[0][s] = __builtin_amdgcn_wmma_f32_16x16x32_f16(
          false, af0, false, bf, (short)0, acc[0][s], false, false);
      acc[1][s] = __builtin_amdgcn_wmma_f32_16x16x32_f16(
          false, af1, false, bf, (short)0, acc[1][s], false, false);
      if (s == 3) { af0 = af0n; af1 = af1n; }
      bf = bfn;
    }
    __syncthreads();

    // ---- commit prefetched chunk into the other buffer
    if (c < 7) {
      _Float16* nxt = ldsx + ((c + 1) & 1) * XTILE;
#pragma unroll
      for (int k = 0; k < NLD; ++k) {
        const int i = tid + k * 256;
        if (i < XLOAD) {
          const int col = i % 66;
          const int r2  = i / 66;
          const int row = r2 % 3;
          const int ic  = r2 / 3;
          nxt[(row * 68 + col) * 40 + ic] = (_Float16)rv[k];
        }
      }
    }
    __syncthreads();
  }

  // ---- epilogue: bias + LeakyReLU(0.2) * sqrt(2); C layout m = j + 8*hi
#pragma unroll
  for (int a = 0; a < 2; ++a) {
    const int ocb = (wave * 2 + a) * 16;
#pragma unroll
    for (int j = 0; j < 8; ++j) {
      const int oc   = ocb + j + 8 * hi;
      const float bb = bias[oc];
      const size_t o = (size_t)(b * 256 + oc) * HW + (size_t)h * WW + w0 + n;
      float v;
      v = acc[a][0][j] + bb; out[o +  0] = (v >= 0.0f ? v : 0.2f * v) * ACT_SCALE;
      v = acc[a][1][j] + bb; out[o + 16] = (v >= 0.0f ? v : 0.2f * v) * ACT_SCALE;
      v = acc[a][2][j] + bb; out[o + 32] = (v >= 0.0f ? v : 0.2f * v) * ACT_SCALE;
      v = acc[a][3][j] + bb; out[o + 48] = (v >= 0.0f ? v : 0.2f * v) * ACT_SCALE;
    }
  }
}

// ---------------------------------------------------------------------------
extern "C" void kernel_launch(void* const* d_in, const int* in_sizes, int n_in,
                              void* d_out, int out_size, void* d_ws, size_t ws_size,
                              hipStream_t stream) {
  const float* x      = (const float*)d_in[0];
  const float* code   = (const float*)d_in[1];
  const float* weight = (const float*)d_in[2];
  const float* bias   = (const float*)d_in[3];
  float* out = (float*)d_out;

  float*    invn = (float*)d_ws;                       // 2048 floats (8 KB)
  _Float16* frag = (_Float16*)((char*)d_ws + 8192);    // 9.4 MB f16 fragments

  modnorm_kernel<<<BB * 256, 256, 0, stream>>>(weight, code, invn);
  modfrag_kernel<<<(FRAG_ELEMS + 255) / 256, 256, 0, stream>>>(weight, code, invn, frag);

  dim3 grid(256, BB, 1);   // strips, batches
  modconv_kernel<<<grid, 256, 0, stream>>>(x, bias, frag, out);
}
